// MNISTQuant_48687749267957
// MI455X (gfx1250) — matrix operand order repaired and checked by reading, not compiled, not run on hardware
//
#include <hip/hip_runtime.h>
#include <hip/hip_bf16.h>
#include <stdint.h>

// ---------------------------------------------------------------------------
// MNIST int8-weight-dequant GEMM for MI455X (gfx1250, wave32, WMMA)
//   out[65536,50] = X[65536,784](f32) x (W_int8[784,50] * scaler)
//
// bf16 WMMA 16x16x32: int8 weights are exact in bf16; scaler applied to the
// f32 accumulator at the end. One wave per 16-row tile, 4 N-tiles wide, so X
// streams from HBM exactly once (~205 MB -> ~9 us floor @ 23.3 TB/s).
//
// Round-2 change: B fragments are staged once per block into LDS via the
// CDNA5 async global->LDS path (ASYNCcnt) instead of being re-fetched from
// L2 every K-step by every wave (cuts fragment traffic 8x and removes the
// s_wait_loadcnt-0-before-every-WMMA serialization).
// ---------------------------------------------------------------------------

typedef __attribute__((ext_vector_type(16))) __bf16 v16bf;
typedef __attribute__((ext_vector_type(8)))  float  v8f;
typedef __attribute__((ext_vector_type(4)))  float  v4f;

#define BATCH   65536
#define IN_F    784
#define OUT_F   50
#define KSTEPS  25                      // ceil(784/32); K zero-padded to 800
#define NTILES  4                       // 50 cols padded to 64 = 4 tiles of 16
#define FRAG_DW (KSTEPS * NTILES * 32 * 8)   // 25600 dwords
#define FRAG_BYTES (FRAG_DW * 4)             // 102,400 B

// ---------------------------------------------------------------------------
// Prep kernel: swizzle int8 W into per-lane bf16 WMMA B-fragments in d_ws.
// Fragment dword index: (((s*NTILES + t)*32 + lane)*8 + v)
// B layout (16-bit, 32x16 KxN, contiguous k order used here, matching A):
//   lane = N + 16*h (h = K half), VGPR v holds K = h*16 + 2v, h*16 + 2v + 1
// ---------------------------------------------------------------------------
__global__ void __launch_bounds__(256)
prep_w_frags(const signed char* __restrict__ w, unsigned int* __restrict__ frags) {
  int idx = blockIdx.x * blockDim.x + threadIdx.x;
  if (idx >= FRAG_DW) return;
  int v    = idx & 7;
  int lane = (idx >> 3) & 31;
  int t    = (idx >> 8) & 3;
  int s    = idx >> 10;

  int n = t * 16 + (lane & 15);
  int h = lane >> 4;
  int k = s * 32 + h * 16 + 2 * v;

  float f0 = 0.0f, f1 = 0.0f;
  if (n < OUT_F) {
    if (k     < IN_F) f0 = (float)w[k       * OUT_F + n];   // int8 exact in bf16
    if (k + 1 < IN_F) f1 = (float)w[(k + 1) * OUT_F + n];
  }
  union { __bf16 h2[2]; unsigned int u; } pk;
  pk.h2[0] = (__bf16)f0;
  pk.h2[1] = (__bf16)f1;
  frags[idx] = pk.u;
}

// ---------------------------------------------------------------------------
// Main kernel.
// A layout (16-bit, 16x32 MxK, contiguous k, as in B): lane = M + 16*h; VGPR v holds
//   K = h*16 + 2v (+1) -> one contiguous 16-float run (two 8-float loads).
// ---------------------------------------------------------------------------
__global__ void __launch_bounds__(256)
mnist_q_wmma(const float* __restrict__ x,
             const unsigned int* __restrict__ frags,
             const float* __restrict__ scaler,
             float* __restrict__ out) {
  extern __shared__ unsigned char smem_raw[];     // 102,400 B of B-fragments

  const int lane   = threadIdx.x & 31;
  const int waveId = blockIdx.x * 8 + (threadIdx.x >> 5);  // 0..4095
  const int m0     = waveId * 16;
  const int h      = lane >> 4;
  const int mRow   = m0 + (lane & 15);

  // ---- Stage B fragments into LDS with CDNA5 async global->LDS copies ----
  // 256 threads x 25 iters x 16 B = 102,400 B; tracked by ASYNCcnt.
  {
    unsigned lds_addr = (unsigned)(uintptr_t)smem_raw + threadIdx.x * 16u;
    const unsigned char* g = (const unsigned char*)frags + threadIdx.x * 16u;
    #pragma unroll
    for (int i = 0; i < FRAG_BYTES / (256 * 16); ++i) {    // 25 iterations
      asm volatile("global_load_async_to_lds_b128 %0, %1, off"
                   :: "v"(lds_addr), "v"(g)
                   : "memory");
      lds_addr += 256 * 16;
      g        += 256 * 16;
    }
    asm volatile("s_wait_asynccnt 0x0" ::: "memory");
  }
  __syncthreads();

  const float* __restrict__ xrow = x + (size_t)mRow * IN_F;
  const v16bf* __restrict__ bl   = (const v16bf*)smem_raw;  // 32 B fragments

  v8f acc[NTILES];
  acc[0] = (v8f){}; acc[1] = (v8f){}; acc[2] = (v8f){}; acc[3] = (v8f){};

  for (int s = 0; s < KSTEPS; ++s) {
    const int kg0 = s * 32 + h * 16;  // 16-aligned; IN_F % 8 == 0 -> all-in/out
    const int kg1 = kg0 + 8;

    // Prefetch next K-slab of this row (speculative; OOB silently dropped)
    __builtin_prefetch(xrow + kg0 + 32, 0, 3);

    v4f a0 = (v4f){}, a1 = (v4f){}, a2 = (v4f){}, a3 = (v4f){};
    if (kg0 < IN_F) { a0 = *(const v4f*)(xrow + kg0); a1 = *(const v4f*)(xrow + kg0 + 4); }
    if (kg1 < IN_F) { a2 = *(const v4f*)(xrow + kg1); a3 = *(const v4f*)(xrow + kg1 + 4); }

    v16bf a;
    a[0]  = (__bf16)a0[0]; a[1]  = (__bf16)a0[1]; a[2]  = (__bf16)a0[2]; a[3]  = (__bf16)a0[3];
    a[4]  = (__bf16)a1[0]; a[5]  = (__bf16)a1[1]; a[6]  = (__bf16)a1[2]; a[7]  = (__bf16)a1[3];
    a[8]  = (__bf16)a2[0]; a[9]  = (__bf16)a2[1]; a[10] = (__bf16)a2[2]; a[11] = (__bf16)a2[3];
    a[12] = (__bf16)a3[0]; a[13] = (__bf16)a3[1]; a[14] = (__bf16)a3[2]; a[15] = (__bf16)a3[3];

    // B fragments from LDS (ds_load_b128 pairs, WGP-local latency)
    const v16bf* __restrict__ bp = bl + (size_t)(s * NTILES) * 32 + lane;
    v16bf b0 = bp[0];
    v16bf b1 = bp[32];
    v16bf b2 = bp[64];
    v16bf b3 = bp[96];

    acc[0] = __builtin_amdgcn_wmma_f32_16x16x32_bf16(false, a, false, b0, (short)0, acc[0], false, false);
    acc[1] = __builtin_amdgcn_wmma_f32_16x16x32_bf16(false, a, false, b1, (short)0, acc[1], false, false);
    acc[2] = __builtin_amdgcn_wmma_f32_16x16x32_bf16(false, a, false, b2, (short)0, acc[2], false, false);
    acc[3] = __builtin_amdgcn_wmma_f32_16x16x32_bf16(false, a, false, b3, (short)0, acc[3], false, false);
  }

  // Epilogue: C/D layout -> row = m0 + h*8 + r, col = t*16 + (lane & 15).
  const float sc   = scaler[0];
  const int   row0 = m0 + h * 8;
  const int   col  = lane & 15;
  #pragma unroll
  for (int t = 0; t < NTILES; ++t) {
    const int c = t * 16 + col;
    if (c < OUT_F) {
      #pragma unroll
      for (int r = 0; r < 8; ++r) {
        out[(size_t)(row0 + r) * OUT_F + c] = acc[t][r] * sc;
      }
    }
  }
}

extern "C" void kernel_launch(void* const* d_in, const int* in_sizes, int n_in,
                              void* d_out, int out_size, void* d_ws, size_t ws_size,
                              hipStream_t stream) {
  const float*       x      = (const float*)d_in[0];
  const signed char* w      = (const signed char*)d_in[1];   // int8 weights
  const float*       scaler = (const float*)d_in[2];
  float*             out    = (float*)d_out;
  unsigned int*      frags  = (unsigned int*)d_ws;           // 102,400 B used

  (void)in_sizes; (void)n_in; (void)out_size; (void)ws_size;

  // Allow 100 KB of dynamic LDS (WGP supports up to 320 KB per workgroup).
  static_assert(FRAG_BYTES == 102400, "fragment table size");
  hipFuncSetAttribute((const void*)mnist_q_wmma,
                      hipFuncAttributeMaxDynamicSharedMemorySize, FRAG_BYTES);

  // 1) Swizzle W into WMMA B-fragment layout (25600 dwords).
  prep_w_frags<<<(FRAG_DW + 255) / 256, 256, 0, stream>>>(w, frags);

  // 2) GEMM: 512 blocks x 8 waves x 16 rows = 65536 rows.
  mnist_q_wmma<<<BATCH / (8 * 16), 256, FRAG_BYTES, stream>>>(x, frags, scaler, out);
}
